// GIN_1168231104920
// MI455X (gfx1250) — matrix operand order repaired and checked by reading, not compiled, not run on hardware
//
#include <hip/hip_runtime.h>
#include <hip/hip_bf16.h>

// ---------------------------------------------------------------------------
// Types for CDNA5 WMMA (gfx1250, wave32)
// ---------------------------------------------------------------------------
typedef __attribute__((ext_vector_type(16))) __bf16 v16bf;
typedef __attribute__((ext_vector_type(8)))  float  v8f;
typedef __attribute__((ext_vector_type(4)))  int    v4i;

#define AS_GLOBAL __attribute__((address_space(1)))
#define AS_LDS    __attribute__((address_space(3)))

union FragBF16 {           // 16 bf16 = 32 bytes
    v16bf v;
    uint4 q[2];
};

#if __has_builtin(__builtin_amdgcn_global_load_async_to_lds_b128) && \
    __has_builtin(__builtin_amdgcn_s_wait_asynccnt)
#define HAS_ASYNC 1
#else
#define HAS_ASYNC 0
#endif

// float -> bf16, round-to-nearest-even
__device__ __forceinline__ unsigned short f2bf(float f) {
    unsigned u = __float_as_uint(f);
    unsigned r = u + 0x7FFFu + ((u >> 16) & 1u);
    return (unsigned short)(r >> 16);
}
__device__ __forceinline__ float bf2f(unsigned short h) {
    return __uint_as_float(((unsigned)h) << 16);
}

// ---------------------------------------------------------------------------
// Problem constants (match reference)
// ---------------------------------------------------------------------------
#define NN  100000
#define DI  256
#define HID 1024
#define NC  64
#define NE  300000

// ---------------------------------------------------------------------------
// Utility kernels
// ---------------------------------------------------------------------------
__global__ void zero_f32_kernel(float* p, size_t n4) {
    size_t stride = (size_t)gridDim.x * blockDim.x;
    for (size_t i = (size_t)blockIdx.x * blockDim.x + threadIdx.x; i < n4; i += stride) {
        ((float4*)p)[i] = make_float4(0.f, 0.f, 0.f, 0.f);
    }
}

__global__ void scatter_f32_kernel(const float* __restrict__ x,
                                   const int* __restrict__ src,
                                   const int* __restrict__ dst,
                                   float* agg, int d4, size_t total) {
    size_t stride = (size_t)gridDim.x * blockDim.x;
    for (size_t idx = (size_t)blockIdx.x * blockDim.x + threadIdx.x; idx < total; idx += stride) {
        int e = (int)(idx / d4);
        int c = (int)(idx - (size_t)e * d4);
        int s = src[e], d = dst[e];
        float4 v = ((const float4*)x)[(size_t)s * d4 + c];
        float* a = agg + (size_t)d * ((size_t)d4 * 4) + (size_t)c * 4;
        atomicAdd(a + 0, v.x);
        atomicAdd(a + 1, v.y);
        atomicAdd(a + 2, v.z);
        atomicAdd(a + 3, v.w);
    }
}

__global__ void scatter_bf16_kernel(const unsigned short* __restrict__ h,
                                    const int* __restrict__ src,
                                    const int* __restrict__ dst,
                                    float* agg, int d4, size_t total) {
    size_t stride = (size_t)gridDim.x * blockDim.x;
    for (size_t idx = (size_t)blockIdx.x * blockDim.x + threadIdx.x; idx < total; idx += stride) {
        int e = (int)(idx / d4);
        int c = (int)(idx - (size_t)e * d4);
        int s = src[e], d = dst[e];
        ushort4 v = ((const ushort4*)h)[(size_t)s * d4 + c];
        float* a = agg + (size_t)d * ((size_t)d4 * 4) + (size_t)c * 4;
        atomicAdd(a + 0, bf2f(v.x));
        atomicAdd(a + 1, bf2f(v.y));
        atomicAdd(a + 2, bf2f(v.z));
        atomicAdd(a + 3, bf2f(v.w));
    }
}

__global__ void combine_f32_kernel(const float* __restrict__ x,
                                   const float* __restrict__ agg,
                                   unsigned short* out, size_t total4) {
    size_t stride = (size_t)gridDim.x * blockDim.x;
    for (size_t i = (size_t)blockIdx.x * blockDim.x + threadIdx.x; i < total4; i += stride) {
        float4 a = ((const float4*)x)[i];
        float4 g = ((const float4*)agg)[i];
        ushort4 o;
        o.x = f2bf(a.x + g.x);
        o.y = f2bf(a.y + g.y);
        o.z = f2bf(a.z + g.z);
        o.w = f2bf(a.w + g.w);
        ((ushort4*)out)[i] = o;
    }
}

__global__ void combine_bf16_kernel(const unsigned short* __restrict__ h,
                                    const float* __restrict__ agg,
                                    unsigned short* out, size_t total4) {
    size_t stride = (size_t)gridDim.x * blockDim.x;
    for (size_t i = (size_t)blockIdx.x * blockDim.x + threadIdx.x; i < total4; i += stride) {
        ushort4 hv = ((const ushort4*)h)[i];
        float4  g  = ((const float4*)agg)[i];
        ushort4 o;
        o.x = f2bf(bf2f(hv.x) + g.x);
        o.y = f2bf(bf2f(hv.y) + g.y);
        o.z = f2bf(bf2f(hv.z) + g.z);
        o.w = f2bf(bf2f(hv.w) + g.w);
        ((ushort4*)out)[i] = o;
    }
}

// Pack fp32 weight W[K][N] (row-major) into B-fragment tiles:
// tile (kt, nt) of shape 32(K) x 16(N) stored as 32 lanes * 16 bf16,
// lane L: N = nt*16 + (L&15), elements j=0..15: K = kt*32 + (L>>4)*16 + j.
__global__ void pack_weight_kernel(const float* __restrict__ W,
                                   unsigned short* out, int K, int N) {
    int ktTotal = K >> 5;
    size_t total = (size_t)K * N;
    size_t stride = (size_t)gridDim.x * blockDim.x;
    for (size_t p = (size_t)blockIdx.x * blockDim.x + threadIdx.x; p < total; p += stride) {
        int tile = (int)(p >> 9);
        int r    = (int)(p & 511);
        int L = r >> 4, j = r & 15;
        int kt = tile % ktTotal;
        int nt = tile / ktTotal;
        int k = (kt << 5) + ((L >> 4) << 4) + j;
        int n = (nt << 4) + (L & 15);
        out[p] = f2bf(W[(size_t)k * N + n]);
    }
}

// ---------------------------------------------------------------------------
// WMMA bf16 GEMM: Out[M][N] = act( A[M][K](bf16,row-major) @ Wpk + bias ).
// Block = 8 waves covering 8 consecutive 16-row M-tiles of ONE N-strip, so
// all waves share the strip's B tiles.  B tiles for each 32-deep k-step are
// staged in LDS (double-buffered) via CDNA5 async-to-LDS copies (ASYNCcnt)
// when available; A fragments are prefetched one k-step ahead.
// ---------------------------------------------------------------------------
template <int NACC, bool RELU, bool OUTF32>
__global__ __launch_bounds__(256)
void gemm_wmma_bf16(const unsigned short* __restrict__ A,
                    const unsigned short* __restrict__ Wpk,
                    const float* __restrict__ bias,
                    void* __restrict__ Out,
                    int M, int K, int N) {
    __shared__ __align__(16) unsigned short ldsB[2][NACC * 512];

    const int tid  = threadIdx.x;
    const int lane = tid & 31;
    const int widb = tid >> 5;

    const int mTiles  = M >> 4;
    const int nStrips = N / (16 * NACC);
    const int strip   = blockIdx.x % nStrips;
    const int mBlock  = blockIdx.x / nStrips;

    int mTile = mBlock * 8 + widb;
    const bool active = (mTile < mTiles);
    if (!active) mTile = mTiles - 1;          // keep wave alive for barriers

    const int m0   = mTile << 4;
    const int nt0  = strip * NACC;
    const int half = lane >> 4;
    const int l15  = lane & 15;
    const int kTiles = K >> 5;

    const unsigned short* Wstrip = Wpk + (size_t)nt0 * kTiles * 512;
    const unsigned short* arow   = A + (size_t)(m0 + l15) * K;

    constexpr int CHUNKS = (NACC * 64) / 256;  // 16B chunks per thread per k-step
    uint4 breg[CHUNKS];                        // fallback staging regs

    auto issueB = [&](int kt, int buf) {
#pragma unroll
        for (int j = 0; j < CHUNKS; ++j) {
            const int c   = tid + j * 256;
            const int i   = c >> 6;          // tile within strip
            const int off = (c & 63) * 8;    // ushort offset within tile
            const unsigned short* g = Wstrip + ((size_t)i * kTiles + kt) * 512 + off;
#if HAS_ASYNC
            __builtin_amdgcn_global_load_async_to_lds_b128(
                (AS_GLOBAL v4i*)g,
                (AS_LDS v4i*)&ldsB[buf][i * 512 + off],
                0, 0);
#else
            breg[j] = *(const uint4*)g;
            (void)buf;
#endif
        }
    };
    auto commitB = [&](int buf) {
#if !HAS_ASYNC
#pragma unroll
        for (int j = 0; j < CHUNKS; ++j) {
            const int c   = tid + j * 256;
            const int i   = c >> 6;
            const int off = (c & 63) * 8;
            *(uint4*)&ldsB[buf][i * 512 + off] = breg[j];
        }
#else
        (void)buf;
#endif
    };
    auto waitAsync = [&]() {
#if HAS_ASYNC
        __builtin_amdgcn_s_wait_asynccnt(0);
#endif
    };

    v8f acc[NACC];
    v8f zero = {};
#pragma unroll
    for (int i = 0; i < NACC; ++i) acc[i] = zero;

    // Prologue: stage k-step 0, prefetch A frag 0.
    issueB(0, 0);
    FragBF16 fa;
    fa.q[0] = *(const uint4*)(arow + (half << 4));
    fa.q[1] = *(const uint4*)(arow + (half << 4) + 8);
    commitB(0);
    waitAsync();
    __syncthreads();

    for (int kt = 0; kt < kTiles; ++kt) {
        const int  cur  = kt & 1;
        const int  nxt  = cur ^ 1;
        const bool more = (kt + 1) < kTiles;

        FragBF16 fan;
        if (more) {
            issueB(kt + 1, nxt);                       // async B for next k-step
            const int kA = ((kt + 1) << 5) + (half << 4);
            fan.q[0] = *(const uint4*)(arow + kA);     // A prefetch
            fan.q[1] = *(const uint4*)(arow + kA + 8);
        }

        // Compute from ldsB[cur]; 2-deep pipeline on B-fragment LDS reads.
        FragBF16 fb0, fb1;
        fb0.q[0] = *(const uint4*)&ldsB[cur][lane * 16];
        fb0.q[1] = *(const uint4*)&ldsB[cur][lane * 16 + 8];
#pragma unroll
        for (int i = 0; i < NACC; ++i) {
            FragBF16& fcur = (i & 1) ? fb1 : fb0;
            FragBF16& fnxt = (i & 1) ? fb0 : fb1;
            if (i + 1 < NACC) {
                fnxt.q[0] = *(const uint4*)&ldsB[cur][(i + 1) * 512 + lane * 16];
                fnxt.q[1] = *(const uint4*)&ldsB[cur][(i + 1) * 512 + lane * 16 + 8];
            }
            acc[i] = __builtin_amdgcn_wmma_f32_16x16x32_bf16(
                false, fa.v, false, fcur.v, (short)0, acc[i], false, false);
        }

        if (more) commitB(nxt);
        waitAsync();
        __syncthreads();
        if (more) fa = fan;
    }

    if (!active) return;

    // Epilogue: bias (+ReLU), store bf16 or f32.
    // C/D layout: lanes 0-15 -> N=lane, M=r; lanes 16-31 -> N=lane-16, M=8+r.
#pragma unroll
    for (int i = 0; i < NACC; ++i) {
        const int n = (nt0 + i) * 16 + l15;
        const float b = bias[n];
#pragma unroll
        for (int r = 0; r < 8; ++r) {
            float v = acc[i][r] + b;
            if (RELU) v = fmaxf(v, 0.0f);
            const int m = m0 + (half << 3) + r;
            if (OUTF32)
                ((float*)Out)[(size_t)m * N + n] = v;
            else
                ((unsigned short*)Out)[(size_t)m * N + n] = f2bf(v);
        }
    }
}

// ---------------------------------------------------------------------------
// In-place log_softmax over 64 classes; one wave per node, 2 classes / lane.
// ---------------------------------------------------------------------------
__global__ __launch_bounds__(256)
void logsoftmax64_kernel(float* out, int nNodes) {
    const int lane = threadIdx.x & 31;
    const int node = blockIdx.x * 8 + (threadIdx.x >> 5);
    if (node >= nNodes) return;
    float2 v = ((const float2*)(out + (size_t)node * 64))[lane];
    float m = fmaxf(v.x, v.y);
#pragma unroll
    for (int off = 16; off > 0; off >>= 1) m = fmaxf(m, __shfl_xor(m, off, 32));
    float s = expf(v.x - m) + expf(v.y - m);
#pragma unroll
    for (int off = 16; off > 0; off >>= 1) s += __shfl_xor(s, off, 32);
    const float l = logf(s);
    float2 r = make_float2(v.x - m - l, v.y - m - l);
    ((float2*)(out + (size_t)node * 64))[lane] = r;
}

// ---------------------------------------------------------------------------
// Launch
// ---------------------------------------------------------------------------
extern "C" void kernel_launch(void* const* d_in, const int* in_sizes, int n_in,
                              void* d_out, int out_size, void* d_ws, size_t ws_size,
                              hipStream_t stream) {
    (void)in_sizes; (void)n_in; (void)out_size; (void)ws_size;

    const float* x   = (const float*)d_in[0];
    const int*   ei  = (const int*)d_in[1];
    const float* W1a = (const float*)d_in[2];
    const float* b1a = (const float*)d_in[3];
    const float* W1b = (const float*)d_in[4];
    const float* b1b = (const float*)d_in[5];
    const float* W2a = (const float*)d_in[6];
    const float* b2a = (const float*)d_in[7];
    const float* W2b = (const float*)d_in[8];
    const float* b2b = (const float*)d_in[9];
    const float* Wfc = (const float*)d_in[10];
    const float* bfc = (const float*)d_in[11];

    const int* src = ei;
    const int* dst = ei + NE;

    // Workspace layout (all offsets 32B aligned)
    char* ws = (char*)d_ws;
    size_t o = 0;
    unsigned short* wpk1a = (unsigned short*)(ws + o); o += (size_t)DI  * HID * 2;
    unsigned short* wpk1b = (unsigned short*)(ws + o); o += (size_t)HID * HID * 2;
    unsigned short* wpk2a = (unsigned short*)(ws + o); o += (size_t)HID * HID * 2;
    unsigned short* wpk2b = (unsigned short*)(ws + o); o += (size_t)HID * HID * 2;
    unsigned short* wpkfc = (unsigned short*)(ws + o); o += (size_t)HID * NC  * 2;
    float*          agg   = (float*)(ws + o);          o += (size_t)NN  * HID * 4;
    unsigned short* bufA  = (unsigned short*)(ws + o); o += (size_t)NN  * HID * 2;
    unsigned short* bufB  = (unsigned short*)(ws + o); o += (size_t)NN  * HID * 2;

    const int T = 256;
    auto gs = [](size_t n) { return (int)((n + 255) / 256); };

    // Pack all weights to bf16 fragment layout (L2-resident, ~7 MB)
    pack_weight_kernel<<<gs((size_t)DI * HID), T, 0, stream>>>(W1a, wpk1a, DI,  HID);
    pack_weight_kernel<<<gs((size_t)HID * HID), T, 0, stream>>>(W1b, wpk1b, HID, HID);
    pack_weight_kernel<<<gs((size_t)HID * HID), T, 0, stream>>>(W2a, wpk2a, HID, HID);
    pack_weight_kernel<<<gs((size_t)HID * HID), T, 0, stream>>>(W2b, wpk2b, HID, HID);
    pack_weight_kernel<<<gs((size_t)HID * NC), T, 0, stream>>>(Wfc, wpkfc, HID, NC);

    const int mTiles  = NN / 16;                 // 6250 (exact)
    const int mBlocks = (mTiles + 7) / 8;        // 782
    const int gemmBlocksH = mBlocks * (HID / 128);   // strips of 128 cols
    const int gemmBlocksC = mBlocks * (NC / 64);     // strips of 64 cols

    // ----- Layer 1 -----
    zero_f32_kernel<<<gs((size_t)NN * DI / 4), T, 0, stream>>>(agg, (size_t)NN * DI / 4);
    scatter_f32_kernel<<<gs((size_t)NE * (DI / 4)), T, 0, stream>>>(
        x, src, dst, agg, DI / 4, (size_t)NE * (DI / 4));
    combine_f32_kernel<<<gs((size_t)NN * DI / 4), T, 0, stream>>>(
        x, agg, bufA, (size_t)NN * DI / 4);
    gemm_wmma_bf16<8, true, false><<<gemmBlocksH, T, 0, stream>>>(
        bufA, wpk1a, b1a, bufB, NN, DI, HID);
    gemm_wmma_bf16<8, true, false><<<gemmBlocksH, T, 0, stream>>>(
        bufB, wpk1b, b1b, bufA, NN, HID, HID);

    // ----- Layer 2 -----
    zero_f32_kernel<<<gs((size_t)NN * HID / 4), T, 0, stream>>>(agg, (size_t)NN * HID / 4);
    scatter_bf16_kernel<<<gs((size_t)NE * (HID / 4)), T, 0, stream>>>(
        bufA, src, dst, agg, HID / 4, (size_t)NE * (HID / 4));
    combine_bf16_kernel<<<gs((size_t)NN * HID / 4), T, 0, stream>>>(
        bufA, agg, bufB, (size_t)NN * HID / 4);
    gemm_wmma_bf16<8, true, false><<<gemmBlocksH, T, 0, stream>>>(
        bufB, wpk2a, b2a, bufA, NN, HID, HID);
    gemm_wmma_bf16<8, false, false><<<gemmBlocksH, T, 0, stream>>>(
        bufA, wpk2b, b2b, bufB, NN, HID, HID);

    // ----- Classifier + log_softmax -----
    gemm_wmma_bf16<4, false, true><<<gemmBlocksC, T, 0, stream>>>(
        bufB, wpkfc, bfc, d_out, NN, HID, NC);
    logsoftmax64_kernel<<<(NN + 7) / 8, T, 0, stream>>>((float*)d_out, NN);
}
